// Interaction_45603962749134
// MI455X (gfx1250) — matrix operand order repaired and checked by reading, not compiled, not run on hardware
//
#include <hip/hip_runtime.h>

// CDNA5 / gfx1250 fused SchNet Interaction block.
// wave32; WMMA f32<-f16 16x16x32 for all GEMMs (f32 accumulate);
// weights pre-packed into B-fragment-major f16 (L2-resident, 32B/lane loads);
// dR_expanded streamed with double-buffered global_load_async_to_lds_b128
// (ASYNCcnt pipeline) overlapping the WMMA work.

typedef __attribute__((ext_vector_type(16))) _Float16 v16h;
typedef __attribute__((ext_vector_type(8)))  _Float16 v8h;
typedef __attribute__((ext_vector_type(8)))  float    v8f;

#define NA   30000
#define NNBH 48
#define NF   128
#define NS   25

// shifted softplus with bare v_exp_f32 / v_log_f32 (arg of log2 is in [1,2]:
// no denorm/inf edge cases, so unguarded HW transcendentals suffice).
// ssp(v) = max(v,0) + ln2*(log2(1 + 2^(-|v|*log2e)) - 1)
__device__ __forceinline__ float ssp_fast(float v) {
  const float t = __builtin_amdgcn_exp2f(-1.4426950408889634f * fabsf(v));
  return fmaxf(v, 0.0f) +
         0.69314718055994531f * (__builtin_amdgcn_logf(1.0f + t) - 1.0f);
}

// ---- WMMA fragment loaders (per CDNA5 ISA 7.12.2 layouts, wave32) ----

// A 16x32 f16 (MxK) from row-major f32, ld=128, 16B-aligned rows, k0 in {0,32,64,96}.
__device__ __forceinline__ v16h load_a_frag_f32_vec(const float* base, int k0, int lane) {
  const int m  = lane & 15;
  const int kb = (lane & 16) ? 8 : 0;
  const float4* row = (const float4*)(base + m * NF);
  const int q = (k0 + kb) >> 2;
  const float4 x0 = row[q];
  const float4 x1 = row[q + 1];
  const float4 x2 = row[q + 4];
  const float4 x3 = row[q + 5];
  v16h a;
  a[0]=(_Float16)x0.x;  a[1]=(_Float16)x0.y;  a[2]=(_Float16)x0.z;  a[3]=(_Float16)x0.w;
  a[4]=(_Float16)x1.x;  a[5]=(_Float16)x1.y;  a[6]=(_Float16)x1.z;  a[7]=(_Float16)x1.w;
  a[8]=(_Float16)x2.x;  a[9]=(_Float16)x2.y;  a[10]=(_Float16)x2.z; a[11]=(_Float16)x2.w;
  a[12]=(_Float16)x3.x; a[13]=(_Float16)x3.y; a[14]=(_Float16)x3.z; a[15]=(_Float16)x3.w;
  return a;
}

// A 16x32 f16 from an LDS-resident f32 tile with ld=25, K zero-padded 25->32.
__device__ __forceinline__ v16h load_a_frag_ldsf32_pad(const float* tile, int lane) {
  const int m  = lane & 15;
  const int kb = (lane & 16) ? 8 : 0;
  const float* row = tile + m * NS;
  v16h a;
#pragma unroll
  for (int i = 0; i < 8; ++i) {
    const int k1 = kb + i;
    const int k2 = 16 + kb + i;
    a[i]     = (k1 < NS) ? (_Float16)row[k1] : (_Float16)0.0f;
    a[i + 8] = (k2 < NS) ? (_Float16)row[k2] : (_Float16)0.0f;
  }
  return a;
}

// A 16x32 f16 from row-major f16 LDS tile (ld=128). Two 16B ds loads per lane.
__device__ __forceinline__ v16h load_a_frag_lds_vec(const _Float16* tile, int k0, int lane) {
  const int m  = lane & 15;
  const int kb = (lane & 16) ? 8 : 0;
  const v8h* row = (const v8h*)(tile + m * NF);
  const v8h lo = row[(k0 + kb) >> 3];
  const v8h hi = row[(k0 + 16 + kb) >> 3];
  v16h a;
#pragma unroll
  for (int i = 0; i < 8; ++i) { a[i] = lo[i]; a[i + 8] = hi[i]; }
  return a;
}

// B 32x16 f16 fragment from packed fragment-major weights: 32B contiguous per lane.
__device__ __forceinline__ v16h load_b_frag_packed(const _Float16* wp, int kt, int nt,
                                                   int lane) {
  const v16h* p = (const v16h*)wp;
  return p[(kt * 8 + nt) * 32 + lane];
}

// Async-copy one 16x25-f32 tile (1600 B = 100 x 16B chunks) global -> LDS.
// Issues exactly 4 global_load_async_to_lds_b128 per wave (ASYNCcnt += 4).
// Generic LDS pointer's low 32 bits are the LDS byte offset (aperture is in [63:32]).
__device__ __forceinline__ void async_prefetch_tile(const float* gsrc, float* ldst,
                                                    int lane) {
  const unsigned ldsoff = (unsigned)(unsigned long long)(uintptr_t)ldst;
  const unsigned long long gaddr = (unsigned long long)(uintptr_t)gsrc;
#pragma unroll
  for (int j = 0; j < 4; ++j) {
    const int chunk = j * 32 + lane;
    if (chunk < 100) {
      asm volatile("global_load_async_to_lds_b128 %0, %1, off"
                   :: "v"(ldsoff + (unsigned)chunk * 16u),
                      "v"(gaddr + (unsigned long long)chunk * 16ull)
                   : "memory");
    }
  }
}

// ---------------------------------------------------------------------------
// One-time weight pack: row-major f32 (Krows x 128, K padded to Kpad) ->
// B-fragment-major f16: dst[((kt*8+nt)*32 + lane)*16 + i].
// ---------------------------------------------------------------------------
__global__ void pack_weights_kernel(const float* __restrict__ W, _Float16* __restrict__ dst,
                                    int Krows, int Kpad) {
  const int idx = blockIdx.x * blockDim.x + threadIdx.x;
  if (idx >= Kpad * NF) return;
  const int k = idx >> 7, n = idx & 127;
  const float v = (k < Krows) ? W[k * NF + n] : 0.0f;
  const int kt = k >> 5, nt = n >> 4;
  const int lane = (n & 15) + ((k & 8) ? 16 : 0);
  const int i = (k & 7) + ((k & 16) ? 8 : 0);
  dst[((kt * 8 + nt) * 32 + lane) * 16 + i] = (_Float16)v;
}

// ---------------------------------------------------------------------------
// Row-tiled GEMM:  out[M,128] = act(A[M,128] @ W[128,128] + bias)
// One wave -> one 16x128 row tile; 4 waves/block; no LDS, no barriers.
// HAS_BIAS / APPLY_SSP are compile-time to avoid per-element uniform branches.
// ---------------------------------------------------------------------------
template <bool HAS_BIAS, bool APPLY_SSP>
__global__ __launch_bounds__(128) void gemm128_rowtile_kernel(
    const float* __restrict__ A, const _Float16* __restrict__ Wp,
    const float* __restrict__ bias, float* __restrict__ out, int M) {
  const int tid  = threadIdx.x;
  const int lane = tid & 31;
  const int wave = tid >> 5;
  const int rt   = blockIdx.x * 4 + wave;
  if (rt >= (M >> 4)) return;

  const float* Arow = A + (size_t)rt * 16 * NF;
  const int fcol = lane & 15;
  const int hi8  = (lane & 16) ? 8 : 0;

  v16h afrag[4];
#pragma unroll
  for (int kt = 0; kt < 4; ++kt)
    afrag[kt] = load_a_frag_f32_vec(Arow, kt * 32, lane);

  float* obase = out + (size_t)(rt * 16 + hi8) * NF;   // row base for this lane half

#pragma unroll
  for (int nt = 0; nt < 8; ++nt) {
    const int f = nt * 16 + fcol;
    const float bval = HAS_BIAS ? bias[f] : 0.0f;
    v8f c;
#pragma unroll
    for (int r = 0; r < 8; ++r) c[r] = bval;
#pragma unroll
    for (int kt = 0; kt < 4; ++kt) {
      v16h b = load_b_frag_packed(Wp, kt, nt, lane);
      c = __builtin_amdgcn_wmma_f32_16x16x32_f16(false, afrag[kt], false, b,
                                                 (short)0, c, false, false);
    }
    float* orow = obase + f;          // stores use constant r*NF immediate offsets
#pragma unroll
    for (int r = 0; r < 8; ++r) {
      float v = c[r];
      if (APPLY_SSP) v = ssp_fast(v);
      orow[r * NF] = v;
    }
  }
}

// ---------------------------------------------------------------------------
// Fused CFConv filter network + neighbor gather + masked aggregation.
// One wave per atom; 4 atoms/block (NA%4==0 -> uniform barriers).
// dRx tiles double-buffered via async global->LDS copies (ASYNCcnt pipeline).
// ---------------------------------------------------------------------------
__global__ __launch_bounds__(128) void cfconv_filter_agg_kernel(
    const float* __restrict__ dRx,          // (NA, 48, 25)
    const int*   __restrict__ neighbors,    // (NA, 48)
    const float* __restrict__ mask,         // (NA, 48)
    const _Float16* __restrict__ Wf1p,      // packed, Kpad=32  (8 frags)
    const float* __restrict__ bf1,          // (128)
    const _Float16* __restrict__ Wf2p,      // packed, Kpad=128 (32 frags)
    const float* __restrict__ bf2,          // (128)
    const float* __restrict__ y,            // (NA,128) f32, L2-resident
    float* __restrict__ agg) {              // (NA,128)
  __shared__ _Float16 htile[4][16 * NF];                // 16 KB, per-wave h staging
  __shared__ __align__(16) float dbuf[4][2][16 * NS];   // 12.8 KB, async dRx tiles
  __shared__ int      nbrrow[4][16];
  __shared__ float    maskrow[4][16];

  const int tid  = threadIdx.x;
  const int lane = tid & 31;
  const int wave = tid >> 5;
  const int a    = blockIdx.x * 4 + wave;   // always < NA
  const int fcol = lane & 15;
  const int hi8  = (lane & 16) ? 8 : 0;

  const float* dsrc = dRx + (size_t)a * NNBH * NS;   // 48*25 floats for this atom

  float aggf[8];
#pragma unroll
  for (int i = 0; i < 8; ++i) aggf[i] = 0.0f;

  // prime the async pipeline with neighbor-tile 0
  async_prefetch_tile(dsrc, &dbuf[wave][0][0], lane);

#pragma unroll
  for (int t = 0; t < 3; ++t) {             // 3 tiles of 16 neighbors
    const int n0 = t * 16;
    if (lane < 16) {
      nbrrow[wave][lane]  = neighbors[a * NNBH + n0 + lane];
      maskrow[wave][lane] = mask[a * NNBH + n0 + lane];
    }

    // issue next tile's async copy, then wait for the current tile only
    if (t < 2) {
      async_prefetch_tile(dsrc + (t + 1) * (16 * NS), &dbuf[wave][(t + 1) & 1][0], lane);
      asm volatile("s_wait_asynccnt 0x4" ::: "memory");
    } else {
      asm volatile("s_wait_asynccnt 0x0" ::: "memory");
    }

    // --- GEMM1: h = ssp(dRx_tile @ Wf1 + bf1), single K step (25 padded to 32) ---
    v16h a1 = load_a_frag_ldsf32_pad(&dbuf[wave][t & 1][0], lane);
    _Float16* hrow = &htile[wave][hi8 * NF];
#pragma unroll
    for (int nt = 0; nt < 8; ++nt) {
      const int f = nt * 16 + fcol;
      const float bval = bf1[f];
      v8f c;
#pragma unroll
      for (int r = 0; r < 8; ++r) c[r] = bval;
      v16h b = load_b_frag_packed(Wf1p, 0, nt, lane);
      c = __builtin_amdgcn_wmma_f32_16x16x32_f16(false, a1, false, b,
                                                 (short)0, c, false, false);
#pragma unroll
      for (int r = 0; r < 8; ++r)
        hrow[r * NF + f] = (_Float16)ssp_fast(c[r]);
    }
    __syncthreads();   // cross-lane visibility of htile / nbr / mask

    // --- GEMM2 (+bias) then gather / mask / aggregate ---
    v16h a2[4];
#pragma unroll
    for (int kt = 0; kt < 4; ++kt)
      a2[kt] = load_a_frag_lds_vec(htile[wave], kt * 32, lane);

#pragma unroll
    for (int nt = 0; nt < 8; ++nt) {
      const int f = nt * 16 + fcol;
      const float bval = bf2[f];
      v8f c;
#pragma unroll
      for (int r = 0; r < 8; ++r) c[r] = bval;
#pragma unroll
      for (int kt = 0; kt < 4; ++kt) {
        v16h b = load_b_frag_packed(Wf2p, kt, nt, lane);
        c = __builtin_amdgcn_wmma_f32_16x16x32_f16(false, a2[kt], false, b,
                                                   (short)0, c, false, false);
      }
      // C/D layout: lane l, vgpr r -> (M=r+hi8, N=f).  Gather L2-resident y rows.
      const float* ycol = y + f;
#pragma unroll
      for (int r = 0; r < 8; ++r) {
        const int m = r + hi8;
        const float yv = ycol[(size_t)nbrrow[wave][m] * NF] * maskrow[wave][m];
        aggf[nt] += c[r] * yv;
      }
    }
    __syncthreads();   // protect htile/nbr/mask against next iteration's writes
  }

  // lanes l and l^16 hold the same column f with disjoint M halves -> pair-reduce
#pragma unroll
  for (int i = 0; i < 8; ++i)
    aggf[i] += __shfl_xor(aggf[i], 16, 32);
  if (lane < 16) {
    float* arow = agg + (size_t)a * NF + lane;
#pragma unroll
    for (int i = 0; i < 8; ++i)
      arow[i * 16] = aggf[i];
  }
}

// ---------------------------------------------------------------------------
extern "C" void kernel_launch(void* const* d_in, const int* in_sizes, int n_in,
                              void* d_out, int out_size, void* d_ws, size_t ws_size,
                              hipStream_t stream) {
  (void)in_sizes; (void)n_in; (void)out_size; (void)ws_size;
  const float* x        = (const float*)d_in[0];
  // d_in[1] = dR (unused by reference math)
  const int*   neighbors= (const int*)  d_in[2];
  const float* pmask    = (const float*)d_in[3];
  const float* dRx      = (const float*)d_in[4];
  const float* Wf1      = (const float*)d_in[5];
  const float* bf1      = (const float*)d_in[6];
  const float* Wf2      = (const float*)d_in[7];
  const float* bf2      = (const float*)d_in[8];
  const float* W_in2f   = (const float*)d_in[9];
  const float* W_f2out  = (const float*)d_in[10];
  const float* b_f2out  = (const float*)d_in[11];
  const float* W_dense  = (const float*)d_in[12];
  const float* b_dense  = (const float*)d_in[13];
  float* out = (float*)d_out;

  // Workspace layout: packed f16 weights first (32B-aligned), then f32 tensors.
  _Float16* hp      = (_Float16*)d_ws;
  _Float16* p_in2f  = hp;                  // 128x128 -> 16384 halves each
  _Float16* p_f2out = hp + 16384;
  _Float16* p_dense = hp + 32768;
  _Float16* p_wf2   = hp + 49152;
  _Float16* p_wf1   = hp + 65536;          // 32x128 -> 4096 halves
  float* fbase = (float*)(hp + 69632);     // 139264 B offset, 16B aligned
  float* y   = fbase;                      // (NA,128)
  float* agg = fbase + (size_t)NA * NF;
  float* tmp = fbase + 2 * (size_t)NA * NF;

  // 0) pack weights into fragment-major f16 (tiny, once per launch)
  pack_weights_kernel<<<64, 256, 0, stream>>>(W_in2f,  p_in2f,  128, 128);
  pack_weights_kernel<<<64, 256, 0, stream>>>(W_f2out, p_f2out, 128, 128);
  pack_weights_kernel<<<64, 256, 0, stream>>>(W_dense, p_dense, 128, 128);
  pack_weights_kernel<<<64, 256, 0, stream>>>(Wf2,     p_wf2,   128, 128);
  pack_weights_kernel<<<16, 256, 0, stream>>>(Wf1,     p_wf1,    25,  32);

  const int rowTiles  = NA / 16;           // 1875
  const int rowBlocks = (rowTiles + 3) / 4;// 469

  // 1) y = x @ W_in2f
  gemm128_rowtile_kernel<false, false><<<rowBlocks, 128, 0, stream>>>(x, p_in2f, nullptr, y, NA);
  // 2) fused filter net + gather + masked neighbor aggregation -> agg
  cfconv_filter_agg_kernel<<<NA / 4, 128, 0, stream>>>(dRx, neighbors, pmask,
                                                       p_wf1, bf1, p_wf2, bf2, y, agg);
  // 3) tmp = ssp(agg @ W_f2out + b_f2out)
  gemm128_rowtile_kernel<true, true><<<rowBlocks, 128, 0, stream>>>(agg, p_f2out, b_f2out, tmp, NA);
  // 4) out = tmp @ W_dense + b_dense
  gemm128_rowtile_kernel<true, false><<<rowBlocks, 128, 0, stream>>>(tmp, p_dense, b_dense, out, NA);
}
